// MixtureOfDepth_27590869909543
// MI455X (gfx1250) — compile-verified, matrix operand order checked by reading, not run on hardware
//
#include <hip/hip_runtime.h>
#include <hip/hip_bf16.h>

typedef __attribute__((ext_vector_type(16))) _Float16 v16h;
typedef __attribute__((ext_vector_type(8)))  float    v8f;

#define Bsz 4
#define Sseq 4096
#define Dm 1024
#define Hh 16
#define HD 64
#define Fm 4096
#define KSEL 512       // top-k
#define KC 511         // kept tokens
#define KPAD 512       // padded rows per batch
#define MROWS (Bsz*KPAD)   // 2048

#define ASTR 40        // A tile LDS row stride (halves): 80B, 16B aligned, bank-skewed
#define BSTR 20        // B tile LDS col stride (dwords): 80B, 16B aligned, bank-skewed
#define QSTR 72        // attention Q LDS row stride (halves)
#define SSTR 520       // attention scores LDS row stride (halves)

// LDS offset of a __shared__ pointer: low 32 bits of the generic address
// (ISA 10.2: LDS aperture maps addr[31:0] directly to the LDS offset).
__device__ __forceinline__ unsigned lds_off(const void* p) {
  return (unsigned)(size_t)p;
}
// CDNA5 async global->LDS copy, 16B per lane, tracked by ASYNCcnt.
__device__ __forceinline__ void async_copy_b128(unsigned lds_dst, const void* gsrc) {
  asm volatile("global_load_async_to_lds_b128 %0, %1, off"
               :: "v"(lds_dst), "v"((unsigned long long)(size_t)gsrc) : "memory");
}
__device__ __forceinline__ void wait_async() {
  asm volatile("s_wait_asynccnt 0x0" ::: "memory");
}

// ---------------------------------------------------------------- utilities
__global__ __launch_bounds__(256) void copy_x_kernel(const float4* __restrict__ src,
                                                     float4* __restrict__ dst, int n4) {
  int i = blockIdx.x * 256 + threadIdx.x;
  if (i < n4) dst[i] = src[i];
}

__global__ __launch_bounds__(256) void cvt_f16_kernel(const float* __restrict__ src,
                                                      _Float16* __restrict__ dst, int n) {
  int i = blockIdx.x * 256 + threadIdx.x;
  if (i < n) dst[i] = (_Float16)src[i];
}

// ---------------------------------------------------------------- router
__global__ __launch_bounds__(256) void router_kernel(const float* __restrict__ x,
                                                     const float* __restrict__ rw,
                                                     const float* __restrict__ rb,
                                                     float* __restrict__ wout) {
  int wave = threadIdx.x >> 5, lane = threadIdx.x & 31;
  int tok = blockIdx.x * 8 + wave;          // < B*S
  const float* xr = x + (size_t)tok * Dm;
  float s = 0.f;
  for (int i = lane; i < Dm; i += 32) s += xr[i] * rw[i];
  for (int m = 16; m >= 1; m >>= 1) s += __shfl_xor(s, m, 32);
  if (lane == 0) wout[tok] = s + rb[0];
}

// ---------------------------------------------------------------- per-batch top-k (bitonic)
__global__ __launch_bounds__(1024) void topk_kernel(const float* __restrict__ wts,
                                                    int* __restrict__ sel_idx,
                                                    float* __restrict__ w_sel) {
  __shared__ float sv[Sseq];
  __shared__ int   si[Sseq];
  __shared__ int   sidx[KSEL];
  int b = blockIdx.x, tid = threadIdx.x;
  const float* w = wts + (size_t)b * Sseq;
  for (int i = tid; i < Sseq; i += 1024) { sv[i] = w[i]; si[i] = i; }
  __syncthreads();
  // sort descending by value, ties -> lower index first (matches lax.top_k)
  for (int k = 2; k <= Sseq; k <<= 1) {
    for (int j = k >> 1; j > 0; j >>= 1) {
      for (int i = tid; i < Sseq; i += 1024) {
        int ixj = i ^ j;
        if (ixj > i) {
          float v0 = sv[i], v1 = sv[ixj];
          int   i0 = si[i], i1 = si[ixj];
          bool dir = ((i & k) == 0);
          bool swap_before = (v1 > v0) || (v1 == v0 && i1 < i0); // arr[ixj] should precede
          if (swap_before == dir) { sv[i] = v1; sv[ixj] = v0; si[i] = i1; si[ixj] = i0; }
        }
      }
      __syncthreads();
    }
  }
  // take top KC indices, sort ascending
  if (tid < KSEL) sidx[tid] = (tid < KC) ? si[tid] : 0x7FFFFFFF;
  __syncthreads();
  for (int k = 2; k <= KSEL; k <<= 1) {
    for (int j = k >> 1; j > 0; j >>= 1) {
      if (tid < KSEL) {
        int i = tid, ixj = i ^ j;
        if (ixj > i) {
          int a = sidx[i], c = sidx[ixj];
          bool dir = ((i & k) == 0);
          if ((c < a) == dir) { sidx[i] = c; sidx[ixj] = a; }
        }
      }
      __syncthreads();
    }
  }
  if (tid < KSEL) {
    int valid = (tid < KC);
    int s = valid ? sidx[tid] : 0;
    sel_idx[b * KPAD + tid] = s;
    w_sel[b * KPAD + tid]   = valid ? w[s] : 0.f;
  }
}

// ---------------------------------------------------------------- gather + rmsnorm -> f16
__global__ __launch_bounds__(256) void gather_rms_kernel(const float* __restrict__ x,
                                                         const int* __restrict__ sel_idx,
                                                         const float* __restrict__ g,
                                                         float* __restrict__ x_sel,
                                                         _Float16* __restrict__ hn) {
  int row = blockIdx.x;          // b*KPAD + j
  int b = row >> 9, j = row & (KPAD - 1);
  int tid = threadIdx.x;
  __shared__ float red[9];
  if (j == KC) {                 // pad row: zeros
    for (int i = tid; i < Dm; i += 256) {
      x_sel[(size_t)row * Dm + i] = 0.f;
      hn[(size_t)row * Dm + i] = (_Float16)0.f;
    }
    return;
  }
  int s = sel_idx[row];
  const float* xr = x + ((size_t)b * Sseq + s) * Dm;
  float vals[4]; float ssq = 0.f;
  for (int q = 0; q < 4; ++q) { float v = xr[tid + q * 256]; vals[q] = v; ssq += v * v; }
  for (int m = 16; m >= 1; m >>= 1) ssq += __shfl_xor(ssq, m, 32);
  if ((tid & 31) == 0) red[tid >> 5] = ssq;
  __syncthreads();
  if (tid == 0) { float t = 0; for (int q = 0; q < 8; ++q) t += red[q];
                  red[8] = rsqrtf(t / (float)Dm + 1e-6f); }
  __syncthreads();
  float sc = red[8];
  for (int q = 0; q < 4; ++q) {
    int c = tid + q * 256;
    x_sel[(size_t)row * Dm + c] = vals[q];
    hn[(size_t)row * Dm + c] = (_Float16)(vals[q] * sc * g[c]);
  }
}

// rmsnorm (no gather) -> f16
__global__ __launch_bounds__(256) void rms_kernel(const float* __restrict__ in,
                                                  const float* __restrict__ g,
                                                  _Float16* __restrict__ outp) {
  int row = blockIdx.x, tid = threadIdx.x;
  __shared__ float red[9];
  const float* xr = in + (size_t)row * Dm;
  float vals[4]; float ssq = 0.f;
  for (int q = 0; q < 4; ++q) { float v = xr[tid + q * 256]; vals[q] = v; ssq += v * v; }
  for (int m = 16; m >= 1; m >>= 1) ssq += __shfl_xor(ssq, m, 32);
  if ((tid & 31) == 0) red[tid >> 5] = ssq;
  __syncthreads();
  if (tid == 0) { float t = 0; for (int q = 0; q < 8; ++q) t += red[q];
                  red[8] = rsqrtf(t / (float)Dm + 1e-6f); }
  __syncthreads();
  float sc = red[8];
  for (int q = 0; q < 4; ++q) {
    int c = tid + q * 256;
    outp[(size_t)row * Dm + c] = (_Float16)(vals[q] * sc * g[c]);
  }
}

// ---------------------------------------------------------------- generic WMMA GEMM
// C[M,N] = A[M,K] * Bw[K,N], A/Bw f16 row-major, f32 accumulate via v_wmma_f32_16x16x32_f16.
// Block: 256 thr (8 waves), 64x128 C tile. Wave (m16 = w&3, nhalf = w>>2) owns 16x64 strip.
// A tile staged via async global->LDS b128 (ASYNCcnt). B tile staged packed as K-pairs,
// transposed in LDS so fragments are two ds_load_b128 each (no v_mov shuffling).
// modes: 0 = store f16 row-major; 1 = store f16 split to [B,H,KPAD,HD] (QKV);
//        2 = store f32 acc + aux (residual); 3 = store f16 gelu(acc).
__global__ __launch_bounds__(256) void gemm_wmma_kernel(const _Float16* __restrict__ A,
                                                        const _Float16* __restrict__ Bw,
                                                        void* __restrict__ C,
                                                        const float* __restrict__ aux,
                                                        int M, int N, int K, int mode) {
  __shared__ _Float16 As[64 * ASTR];    // 5 KB
  __shared__ unsigned Bs[128 * BSTR];   // 10 KB, Bs[n*BSTR + k2] = packed (2k2, 2k2+1) pair
  int tid = threadIdx.x;
  int wave = tid >> 5, lane = tid & 31;
  int lrow = lane & 15, lhi = lane >> 4;
  int ntiles = N >> 7;
  int mbase = (blockIdx.x / ntiles) * 64;
  int nbase = (blockIdx.x % ntiles) * 128;
  int m16 = wave & 3, nhalf = wave >> 2;
  v8f acc[4] = {};
  // per-thread staging coordinates
  int arow_ld = tid >> 2, acg = (tid & 3) * 8;           // A: 16B per thread
  unsigned a_lds = lds_off(&As[arow_ld * ASTR + acg]);
  int bn = tid & 127, k2b = (tid >> 7) * 8;              // B: 8 packed dwords per thread
  for (int kb = 0; kb < K; kb += 32) {
    { // async stage A 64x32
      const _Float16* src = A + (size_t)(mbase + arow_ld) * K + kb + acg;
      async_copy_b128(a_lds, src);
      if (kb + 32 < K) __builtin_prefetch(src + 32, 0, 3);
    }
    { // stage B 32x128 packed as (k even, k odd) pairs, transposed layout
      const _Float16* p0 = Bw + (size_t)(kb + 2 * k2b) * N + nbase + bn;
      if (kb + 32 < K) __builtin_prefetch(p0 + (size_t)32 * N, 0, 3);
      for (int q = 0; q < 8; ++q) {
        const _Float16* p = p0 + (size_t)(2 * q) * N;
        unsigned lo = *(const unsigned short*)p;
        unsigned hi = *(const unsigned short*)(p + N);
        Bs[bn * BSTR + k2b + q] = (hi << 16) | lo;
      }
    }
    wait_async();
    __syncthreads();
    union { v16h v; uint4 q[2]; } af;
    int ar = m16 * 16 + lrow;
    af.q[0] = *(const uint4*)&As[ar * ASTR + 8 * lhi];
    af.q[1] = *(const uint4*)&As[ar * ASTR + 8 * lhi + 16];
    #pragma unroll
    for (int t = 0; t < 4; ++t) {
      union { v16h v; uint4 q[2]; } bf;
      int ncol = nhalf * 64 + t * 16 + lrow;
      bf.q[0] = *(const uint4*)&Bs[ncol * BSTR + 4 * lhi];
      bf.q[1] = *(const uint4*)&Bs[ncol * BSTR + 4 * lhi + 8];
      acc[t] = __builtin_amdgcn_wmma_f32_16x16x32_f16(false, af.v, false, bf.v,
                                                      (short)0, acc[t], false, false);
    }
    __syncthreads();
  }
  #pragma unroll
  for (int t = 0; t < 4; ++t) {
    #pragma unroll
    for (int r = 0; r < 8; ++r) {
      int row = mbase + m16 * 16 + r + 8 * lhi;
      int col = nbase + nhalf * 64 + t * 16 + lrow;
      float val = acc[t][r];
      size_t off = (size_t)row * N + col;
      if (mode == 0) {
        ((_Float16*)C)[off] = (_Float16)val;
      } else if (mode == 1) {            // split heads: row = b*KPAD+j, col = h*HD+t
        int b = row >> 9, jj = row & (KPAD - 1);
        int h = col >> 6, td = col & (HD - 1);
        ((_Float16*)C)[(((size_t)(b * Hh + h) * KPAD + jj) << 6) + td] = (_Float16)val;
      } else if (mode == 2) {
        ((float*)C)[off] = val + aux[off];
      } else {                           // tanh-approx gelu
        float u = val;
        float gg = 0.5f * u * (1.0f + tanhf(0.7978845608028654f * (u + 0.044715f * u * u * u)));
        ((_Float16*)C)[off] = (_Float16)gg;
      }
    }
  }
}

// ---------------------------------------------------------------- RoPE on q,k [B,H,KPAD,HD]
__global__ __launch_bounds__(256) void rope_kernel(_Float16* __restrict__ q,
                                                   _Float16* __restrict__ kk,
                                                   const int* __restrict__ sel_idx,
                                                   const int* __restrict__ pos_ids) {
  int t = blockIdx.x * 256 + threadIdx.x;      // over B*H*KPAD*32 pairs
  if (t >= Bsz * Hh * KPAD * 32) return;
  int i = t & 31;
  int rest = t >> 5;
  int j = rest & (KPAD - 1);
  int bh = rest >> 9;
  int b = bh >> 4;
  int s = sel_idx[b * KPAD + j];
  float pos = (float)pos_ids[b * Sseq + s];
  float inv = __powf(10000.0f, -(float)i / 32.0f);
  float f = pos * inv, cs = __cosf(f), sn = __sinf(f);
  size_t base = ((size_t)bh * KPAD + j) * HD;
  float q0 = (float)q[base + i], q1 = (float)q[base + i + 32];
  q[base + i]      = (_Float16)(q0 * cs - q1 * sn);
  q[base + i + 32] = (_Float16)(q1 * cs + q0 * sn);
  float k0 = (float)kk[base + i], k1 = (float)kk[base + i + 32];
  kk[base + i]      = (_Float16)(k0 * cs - k1 * sn);
  kk[base + i + 32] = (_Float16)(k1 * cs + k0 * sn);
}

// ---------------------------------------------------------------- fused attention
// One block = one (b,h) and 32 query rows. Causal mask is implicit: sel_idx is sorted
// ascending, so the gathered mask equals plain causal on the compact sequence -> the
// 256MB attention_mask input is never read.
// scores -> f16 LDS -> softmax -> P*V, all matmuls via WMMA.
__global__ __launch_bounds__(256) void attn_kernel(const _Float16* __restrict__ qg,
                                                   const _Float16* __restrict__ kg,
                                                   const _Float16* __restrict__ vg,
                                                   _Float16* __restrict__ og) {
  __shared__ _Float16 qs[32 * QSTR];    // 4.5 KB
  __shared__ _Float16 ss[32 * SSTR];    // 32.5 KB scores/probs (row stride skews banks)
  int tid = threadIdx.x, wave = tid >> 5, lane = tid & 31;
  int lrow = lane & 15, lhi = lane >> 4;
  int bh = blockIdx.x >> 4, mblk = blockIdx.x & 15;
  int b = bh >> 4, h = bh & 15;
  const _Float16* qb = qg + ((size_t)bh * KPAD + mblk * 32) * HD;
  const _Float16* kb = kg + (size_t)bh * KPAD * HD;
  const _Float16* vb = vg + (size_t)bh * KPAD * HD;
  { // async stage Q 32x64
    int row = tid >> 3, cg = (tid & 7) * 8;
    async_copy_b128(lds_off(&qs[row * QSTR + cg]), &qb[row * HD + cg]);
    wait_async();
  }
  __syncthreads();
  { // scores: wave -> m16 = w&1, key tiles 8*(w>>1) .. +7
    int m16 = wave & 1;
    union { v16h v; uint4 q[2]; } a0, a1;
    int ar = m16 * 16 + lrow;
    a0.q[0] = *(const uint4*)&qs[ar * QSTR + 8 * lhi];
    a0.q[1] = *(const uint4*)&qs[ar * QSTR + 8 * lhi + 16];
    a1.q[0] = *(const uint4*)&qs[ar * QSTR + 8 * lhi + 32];
    a1.q[1] = *(const uint4*)&qs[ar * QSTR + 8 * lhi + 48];
    int nt0 = (wave >> 1) * 8;
    for (int nt = nt0; nt < nt0 + 8; ++nt) {
      union { v16h v; unsigned u[8]; } b0, b1;
      int key = nt * 16 + lrow;
      #pragma unroll
      for (int v = 0; v < 8; ++v) {
        int kk0 = 8 * lhi + 2 * (v & 3) + 16 * (v >> 2);
        b0.u[v] = *(const unsigned*)&kb[(size_t)key * HD + kk0];
        b1.u[v] = *(const unsigned*)&kb[(size_t)key * HD + 32 + kk0];
      }
      v8f acc = {};
      acc = __builtin_amdgcn_wmma_f32_16x16x32_f16(false, a0.v, false, b0.v, (short)0, acc, false, false);
      acc = __builtin_amdgcn_wmma_f32_16x16x32_f16(false, a1.v, false, b1.v, (short)0, acc, false, false);
      #pragma unroll
      for (int r = 0; r < 8; ++r) {
        int rr = m16 * 16 + r + 8 * lhi;
        ss[rr * SSTR + nt * 16 + lrow] = (_Float16)(acc[r] * 0.125f); // 1/sqrt(64)
      }
    }
  }
  __syncthreads();
  { // causal softmax (valid cols: c <= q_global, c < KC)
    for (int rr = wave * 4; rr < wave * 4 + 4; ++rr) {
      int jq = mblk * 32 + rr;
      int lim = (jq < KC ? jq : KC - 1);
      float mx = -1e30f;
      for (int c = lane; c <= lim; c += 32) mx = fmaxf(mx, (float)ss[rr * SSTR + c]);
      for (int m = 16; m >= 1; m >>= 1) mx = fmaxf(mx, __shfl_xor(mx, m, 32));
      float sm = 0.f;
      for (int c = lane; c <= lim; c += 32) sm += __expf((float)ss[rr * SSTR + c] - mx);
      for (int m = 16; m >= 1; m >>= 1) sm += __shfl_xor(sm, m, 32);
      float inv = 1.0f / sm;
      for (int c = lane; c < KPAD; c += 32) {
        float p = (c <= lim) ? __expf((float)ss[rr * SSTR + c] - mx) * inv : 0.f;
        ss[rr * SSTR + c] = (_Float16)p;
      }
    }
  }
  __syncthreads();
  { // out = P(32x512) * V(512x64): wave -> m16 = w&1, hd tile = w>>1
    int m16 = wave & 1, ntile = wave >> 1;
    v8f acc = {};
    for (int kc2 = 0; kc2 < KPAD / 32; ++kc2) {
      union { v16h v; uint4 q[2]; } af;
      int ar = m16 * 16 + lrow;
      af.q[0] = *(const uint4*)&ss[ar * SSTR + kc2 * 32 + 8 * lhi];
      af.q[1] = *(const uint4*)&ss[ar * SSTR + kc2 * 32 + 8 * lhi + 16];
      union { v16h v; unsigned u[8]; } bf;
      int col = ntile * 16 + lrow;
      #pragma unroll
      for (int v = 0; v < 8; ++v) {
        int kk0 = kc2 * 32 + 8 * lhi + 2 * (v & 3) + 16 * (v >> 2);
        unsigned lo = *(const unsigned short*)&vb[(size_t)kk0 * HD + col];
        unsigned hi = *(const unsigned short*)&vb[(size_t)(kk0 + 1) * HD + col];
        bf.u[v] = (hi << 16) | lo;
      }
      acc = __builtin_amdgcn_wmma_f32_16x16x32_f16(false, af.v, false, bf.v, (short)0, acc, false, false);
    }
    #pragma unroll
    for (int r = 0; r < 8; ++r) {
      int j = mblk * 32 + m16 * 16 + r + 8 * lhi;
      int d = h * HD + ntile * 16 + lrow;
      og[((size_t)(b * KPAD + j)) * Dm + d] = (_Float16)acc[r];
    }
  }
}

// ---------------------------------------------------------------- final scatter
__global__ __launch_bounds__(256) void scatter_kernel(const float* __restrict__ x2,
                                                      const int* __restrict__ sel_idx,
                                                      const float* __restrict__ w_sel,
                                                      float* __restrict__ out) {
  int row = blockIdx.x;          // b*KPAD + j
  int b = row >> 9, j = row & (KPAD - 1);
  if (j >= KC) return;
  int s = sel_idx[row];
  float wv = w_sel[row];
  const float* src = x2 + (size_t)row * Dm;
  float* dst = out + ((size_t)b * Sseq + s) * Dm;
  for (int i = threadIdx.x; i < Dm; i += 256) dst[i] = src[i] * wv;
}

// ---------------------------------------------------------------- launch
extern "C" void kernel_launch(void* const* d_in, const int* in_sizes, int n_in,
                              void* d_out, int out_size, void* d_ws, size_t ws_size,
                              hipStream_t stream) {
  const float* x        = (const float*)d_in[0];
  // d_in[1] attention_mask: unused (sorted sel_idx => implicit causal mask)
  const int*   pos_ids  = (const int*)d_in[2];
  const float* router_w = (const float*)d_in[3];
  const float* router_b = (const float*)d_in[4];
  const float* wq = (const float*)d_in[5];
  const float* wk = (const float*)d_in[6];
  const float* wv = (const float*)d_in[7];
  const float* wo = (const float*)d_in[8];
  const float* w1 = (const float*)d_in[9];
  const float* w2 = (const float*)d_in[10];
  const float* ln1 = (const float*)d_in[11];
  const float* ln2 = (const float*)d_in[12];
  float* out = (float*)d_out;

  char* ws = (char*)d_ws;
  size_t off = 0;
  auto alloc = [&](size_t bytes) -> char* {
    char* p = ws + off;
    off += (bytes + 255) & ~(size_t)255;
    return p;
  };
  float*    wts   = (float*)alloc((size_t)Bsz * Sseq * 4);
  int*      sel   = (int*)alloc((size_t)Bsz * KPAD * 4);
  float*    wsel  = (float*)alloc((size_t)Bsz * KPAD * 4);
  float*    x_sel = (float*)alloc((size_t)MROWS * Dm * 4);
  _Float16* hn    = (_Float16*)alloc((size_t)MROWS * Dm * 2);
  _Float16* wq16  = (_Float16*)alloc((size_t)Dm * Dm * 2);
  _Float16* wk16  = (_Float16*)alloc((size_t)Dm * Dm * 2);
  _Float16* wv16  = (_Float16*)alloc((size_t)Dm * Dm * 2);
  _Float16* wo16  = (_Float16*)alloc((size_t)Dm * Dm * 2);
  _Float16* w1h   = (_Float16*)alloc((size_t)Dm * Fm * 2);
  _Float16* w2h   = (_Float16*)alloc((size_t)Fm * Dm * 2);
  _Float16* qh    = (_Float16*)alloc((size_t)MROWS * Dm * 2);
  _Float16* kh    = (_Float16*)alloc((size_t)MROWS * Dm * 2);
  _Float16* vh    = (_Float16*)alloc((size_t)MROWS * Dm * 2);
  _Float16* attn  = (_Float16*)alloc((size_t)MROWS * Dm * 2);
  float*    x1    = (float*)alloc((size_t)MROWS * Dm * 4);
  _Float16* h2    = (_Float16*)alloc((size_t)MROWS * Dm * 2);
  _Float16* mid   = (_Float16*)alloc((size_t)MROWS * Fm * 2);
  float*    x2    = (float*)alloc((size_t)MROWS * Dm * 4);

  int nElem = Bsz * Sseq * Dm;
  copy_x_kernel<<<nElem / 4 / 256, 256, 0, stream>>>((const float4*)x, (float4*)out, nElem / 4);
  router_kernel<<<(Bsz * Sseq) / 8, 256, 0, stream>>>(x, router_w, router_b, wts);
  topk_kernel<<<Bsz, 1024, 0, stream>>>(wts, sel, wsel);
  gather_rms_kernel<<<MROWS, 256, 0, stream>>>(x, sel, ln1, x_sel, hn);

  cvt_f16_kernel<<<(Dm * Dm) / 256, 256, 0, stream>>>(wq, wq16, Dm * Dm);
  cvt_f16_kernel<<<(Dm * Dm) / 256, 256, 0, stream>>>(wk, wk16, Dm * Dm);
  cvt_f16_kernel<<<(Dm * Dm) / 256, 256, 0, stream>>>(wv, wv16, Dm * Dm);
  cvt_f16_kernel<<<(Dm * Dm) / 256, 256, 0, stream>>>(wo, wo16, Dm * Dm);
  cvt_f16_kernel<<<(Dm * Fm) / 256, 256, 0, stream>>>(w1, w1h, Dm * Fm);
  cvt_f16_kernel<<<(Fm * Dm) / 256, 256, 0, stream>>>(w2, w2h, Fm * Dm);

  int gProj = (MROWS / 64) * (Dm / 128);   // 256 blocks
  gemm_wmma_kernel<<<gProj, 256, 0, stream>>>(hn, wq16, qh, nullptr, MROWS, Dm, Dm, 1);
  gemm_wmma_kernel<<<gProj, 256, 0, stream>>>(hn, wk16, kh, nullptr, MROWS, Dm, Dm, 1);
  gemm_wmma_kernel<<<gProj, 256, 0, stream>>>(hn, wv16, vh, nullptr, MROWS, Dm, Dm, 1);

  rope_kernel<<<(Bsz * Hh * KPAD * 32) / 256, 256, 0, stream>>>(qh, kh, sel, pos_ids);

  attn_kernel<<<Bsz * Hh * (KPAD / 32), 256, 0, stream>>>(qh, kh, vh, attn);

  gemm_wmma_kernel<<<gProj, 256, 0, stream>>>(attn, wo16, x1, x_sel, MROWS, Dm, Dm, 2);
  rms_kernel<<<MROWS, 256, 0, stream>>>(x1, ln2, h2);

  int gMlp1 = (MROWS / 64) * (Fm / 128);   // 1024 blocks
  gemm_wmma_kernel<<<gMlp1, 256, 0, stream>>>(h2, w1h, mid, nullptr, MROWS, Fm, Dm, 3);
  gemm_wmma_kernel<<<gProj, 256, 0, stream>>>(mid, w2h, x2, x1, MROWS, Dm, Fm, 2);

  scatter_kernel<<<MROWS, 256, 0, stream>>>(x2, sel, wsel, out);
  (void)in_sizes; (void)n_in; (void)out_size; (void)ws_size;
}
